// GNNLayer_37641093382252
// MI455X (gfx1250) — matrix operand order repaired and checked by reading, not compiled, not run on hardware
//
#include <hip/hip_runtime.h>

// ---------------------------------------------------------------------------
// GNN bipartite layer for MI455X (gfx1250, wave32, WMMA).
//   Phase A: scatter-aggregate  lit -> cls   (mean ch 0..63, max ch 64..127)
//   Phase B: fused concat + MLP(256->128 SiLU ->128) + residual + LN  (WMMA)
//   Phase C: scatter-aggregate  cls_new -> lit
//   Phase D: fused concat(+flip) + MLP(384->128->128) + residual + LN (WMMA)
//
// Roofline: ~47 GFLOP of GEMM (trivial for WMMA f16->f32) vs ~2 GB of
// L2-resident gather + 512M f32 atomics in the scatter phases -> scatter is
// the bound; node phases are fully fused so z/hidden never touch HBM.
// ---------------------------------------------------------------------------

typedef __attribute__((ext_vector_type(16))) _Float16 v16h;
typedef __attribute__((ext_vector_type(8)))  _Float16 v8h;
typedef __attribute__((ext_vector_type(8)))  float    v8f;

#define N_LIT 262144
#define N_CLS 131072
#define NEDGE 2097152
#define C     128
#define EPS   1e-5f

// float atomic max via monotone int/uint reinterpretation (no NaNs expected)
__device__ __forceinline__ void atomicMaxF(float* addr, float v) {
    if (v >= 0.0f) atomicMax((int*)addr, __float_as_int(v));
    else           atomicMin((unsigned int*)addr, __float_as_uint(v));
}

// Vectorized fill: 4 floats per thread (n must be a multiple of 4 -> true here)
__global__ __launch_bounds__(256) void fill_f32x4(float4* __restrict__ p, float v, int n4) {
    int i = blockIdx.x * 256 + threadIdx.x;
    if (i < n4) p[i] = make_float4(v, v, v, v);
}

// 32 threads per edge, float4 per thread: one global_load_b128 gather + 4
// atomics each; a full wave covers one 512B source row. Index loads amortized
// 4x vs scalar version. L2-atomic-throughput bound by design.
__global__ __launch_bounds__(256) void scatter_agg(
    const float* __restrict__ x, const int* __restrict__ src_idx,
    const int* __restrict__ dst_idx,
    float* __restrict__ sumB, float* __restrict__ maxB, float* __restrict__ degB)
{
    long long idx = (long long)blockIdx.x * 256 + threadIdx.x;
    int e = (int)(idx >> 5);          // edge id
    int q = (int)(idx & 31);          // quad id: channels 4q..4q+3
    if (e >= NEDGE) return;
    int s = src_idx[e];
    int d = dst_idx[e];
    const float4 v = *(const float4*)&x[(long long)s * C + 4 * q];
    if (q < 16) {                     // channels 0..63 -> mean (sum)
        float* dst = &sumB[(long long)d * 64 + 4 * q];
        atomicAdd(dst + 0, v.x);
        atomicAdd(dst + 1, v.y);
        atomicAdd(dst + 2, v.z);
        atomicAdd(dst + 3, v.w);
    } else {                          // channels 64..127 -> max
        float* dst = &maxB[(long long)d * 64 + 4 * (q - 16)];
        atomicMaxF(dst + 0, v.x);
        atomicMaxF(dst + 1, v.y);
        atomicMaxF(dst + 2, v.z);
        atomicMaxF(dst + 3, v.w);
    }
    if (q == 0) atomicAdd(&degB[d], 1.0f);
}

// Pack a (K x 128) f32 row-major weight matrix into fragment-contiguous f16
// B-operands for v_wmma_f32_16x16x32_f16:
//   dst[((nt*KT + kt)*32 + lane)*16 + e] = W[kt*32 + (e>>3)*16 + (lane>>4)*8 + (e&7)][nt*16 + (lane&15)]
__global__ __launch_bounds__(256) void pack_w(const float* __restrict__ w,
                                              _Float16* __restrict__ dst, int K)
{
    int total = 8 * (K / 32) * 512;
    int i = blockIdx.x * 256 + threadIdx.x;
    if (i >= total) return;
    int KT   = K / 32;
    int e    = i & 15;
    int lane = (i >> 4) & 31;
    int kt   = (i >> 9) % KT;
    int nt   = (i >> 9) / KT;
    int kk   = kt * 32 + (lane >> 4) * 8 + e + ((e >> 3) << 3);
    int col  = nt * 16 + (lane & 15);
    dst[i] = (_Float16)w[kk * C + col];
}

// Fused per-node kernel: block = 16 nodes, 256 threads = 8 waves.
// Wave w owns output columns [16w, 16w+16). KZ = 256 (cls) or 384 (lit).
template <int KZ, bool IS_CLS>
__global__ __launch_bounds__(256) void mlp_ln(
    const float* __restrict__ h_in,                // (N x 128) residual/source
    const float* __restrict__ sumB, const float* __restrict__ maxB,
    const float* __restrict__ degB,
    const _Float16* __restrict__ w1p, const _Float16* __restrict__ w2p,
    const float* __restrict__ b1, const float* __restrict__ b2,
    const float* __restrict__ gamma, const float* __restrict__ beta,
    float* __restrict__ out)
{
    constexpr int KT1 = KZ / 32;          // k-tiles, layer 1
    constexpr int ZP  = KZ + 8;           // padded z row (f16) to spread banks
    constexpr int HP  = C + 8;            // padded hidden row (f16)

    __shared__ __align__(32) _Float16 zs[16 * ZP];
    __shared__ __align__(32) _Float16 hs[16 * HP];
    __shared__ float outs[16 * C];
    __shared__ float redS[16 * 16];
    __shared__ float redQ[16 * 16];
    __shared__ float muS[16], rsS[16];

    const int t    = threadIdx.x;
    const int lane = t & 31;
    const int wid  = t >> 5;
    const long long rowBase = (long long)blockIdx.x * 16;

    // ---- build z tile (f16) in LDS: [h | (flip) | mean | masked-max] ----
    {
        int r  = t >> 4;                  // node row 0..15
        int cw = t & 15;                  // chunk worker
        long long node = rowBase + r;
        for (int ch = cw; ch < KZ / 16; ch += 16) {
            float dg = 0.0f;
            #pragma unroll
            for (int u = 0; u < 16; ++u) {
                int k = ch * 16 + u;
                float val;
                if (IS_CLS) {
                    if (k < C) {
                        val = h_in[node * C + k];
                    } else if (k < C + 64) {
                        dg  = degB[node];
                        val = sumB[node * 64 + (k - C)] / fmaxf(dg, 1.0f);
                    } else {
                        dg  = degB[node];
                        val = (dg > 0.0f) ? maxB[node * 64 + (k - C - 64)] : 0.0f;
                    }
                } else {
                    if (k < C) {
                        val = h_in[node * C + k];
                    } else if (k < 2 * C) {
                        val = h_in[(node ^ 1) * C + (k - C)];   // pair flip
                    } else if (k < 2 * C + 64) {
                        dg  = degB[node];
                        val = sumB[node * 64 + (k - 2 * C)] / fmaxf(dg, 1.0f);
                    } else {
                        dg  = degB[node];
                        val = (dg > 0.0f) ? maxB[node * 64 + (k - 2 * C - 64)] : 0.0f;
                    }
                }
                zs[r * ZP + k] = (_Float16)val;
            }
        }
    }
    __syncthreads();

    const int r = lane & 15;              // A-fragment row for this lane
    const int h = lane >> 4;              // half selector (K sub-stripe)

    union AB { v16h v; struct { v8h lo; v8h hi; } p; };

    // ---- GEMM1: hidden = silu(z @ W1 + b1), per-wave 16x16 tile ----
    v8f acc = {};
    #pragma unroll
    for (int kt = 0; kt < KT1; ++kt) {
        AB a;
        a.p.lo = *(const v8h*)&zs[r * ZP + kt * 32 + h * 8];        // K = g0
        a.p.hi = *(const v8h*)&zs[r * ZP + kt * 32 + 16 + h * 8];   // K = g1
        v16h b = *(const v16h*)&w1p[((wid * KT1 + kt) * 32 + lane) * 16];
        acc = __builtin_amdgcn_wmma_f32_16x16x32_f16(
            false, a.v, false, b, (short)0, acc, false, false);
    }
    {
        int col = wid * 16 + (lane & 15);
        float bb = b1[col];
        #pragma unroll
        for (int v = 0; v < 8; ++v) {
            int m = v + 8 * (lane >> 4);
            float x = acc[v] + bb;
            float s = x / (1.0f + __expf(-x));                      // SiLU
            hs[m * HP + col] = (_Float16)s;
        }
    }
    __syncthreads();

    // ---- GEMM2: out = hidden @ W2 + b2 + residual ----
    v8f acc2 = {};
    #pragma unroll
    for (int kt = 0; kt < 4; ++kt) {
        AB a;
        a.p.lo = *(const v8h*)&hs[r * HP + kt * 32 + h * 8];
        a.p.hi = *(const v8h*)&hs[r * HP + kt * 32 + 16 + h * 8];
        v16h b = *(const v16h*)&w2p[((wid * 4 + kt) * 32 + lane) * 16];
        acc2 = __builtin_amdgcn_wmma_f32_16x16x32_f16(
            false, a.v, false, b, (short)0, acc2, false, false);
    }
    {
        int col = wid * 16 + (lane & 15);
        float bb = b2[col];
        #pragma unroll
        for (int v = 0; v < 8; ++v) {
            int m = v + 8 * (lane >> 4);
            long long node = rowBase + m;
            outs[m * C + col] = acc2[v] + bb + h_in[node * C + col];
        }
    }
    __syncthreads();

    // ---- LayerNorm over 128 channels, 16 workers per row ----
    {
        int rr = t >> 4;
        int p  = t & 15;
        float s = 0.0f, q = 0.0f;
        #pragma unroll
        for (int u = 0; u < 8; ++u) {
            float x = outs[rr * C + p * 8 + u];
            s += x; q += x * x;
        }
        redS[rr * 16 + p] = s;
        redQ[rr * 16 + p] = q;
    }
    __syncthreads();
    if (t < 16) {
        float s = 0.0f, q = 0.0f;
        #pragma unroll
        for (int p = 0; p < 16; ++p) { s += redS[t * 16 + p]; q += redQ[t * 16 + p]; }
        float mu  = s * (1.0f / C);
        float var = q * (1.0f / C) - mu * mu;
        muS[t] = mu;
        rsS[t] = rsqrtf(var + EPS);
    }
    __syncthreads();
    {
        int rr = t >> 4;
        int p  = t & 15;
        long long node = rowBase + rr;
        float mu = muS[rr], rs = rsS[rr];
        #pragma unroll
        for (int u = 0; u < 8; ++u) {
            int k = p * 8 + u;
            float x = outs[rr * C + k];
            out[node * C + k] = (x - mu) * rs * gamma[k] + beta[k];
        }
    }
}

extern "C" void kernel_launch(void* const* d_in, const int* in_sizes, int n_in,
                              void* d_out, int out_size, void* d_ws, size_t ws_size,
                              hipStream_t stream)
{
    (void)in_sizes; (void)n_in; (void)out_size; (void)ws_size;

    const float* h_lit     = (const float*)d_in[0];
    const float* h_cls     = (const float*)d_in[1];
    const int*   edge_cls  = (const int*)d_in[2];
    const int*   edge_lit  = (const int*)d_in[3];
    const float* cls_w1    = (const float*)d_in[4];
    const float* cls_b1    = (const float*)d_in[5];
    const float* cls_w2    = (const float*)d_in[6];
    const float* cls_b2    = (const float*)d_in[7];
    const float* lit_w1    = (const float*)d_in[8];
    const float* lit_b1    = (const float*)d_in[9];
    const float* lit_w2    = (const float*)d_in[10];
    const float* lit_b2    = (const float*)d_in[11];
    const float* cls_gamma = (const float*)d_in[12];
    const float* cls_beta  = (const float*)d_in[13];
    const float* lit_gamma = (const float*)d_in[14];
    const float* lit_beta  = (const float*)d_in[15];

    float* out_lit = (float*)d_out;                       // (N_LIT x 128)
    float* out_cls = out_lit + (size_t)N_LIT * C;         // (N_CLS x 128)

    // Workspace layout (sized for the larger lit phase; reused by cls phase):
    //   sumB : N_LIT*64 f32   maxB : N_LIT*64 f32   degB : N_LIT f32
    //   then fragment-packed f16 weights.
    float*     sumB = (float*)d_ws;
    float*     maxB = sumB + (size_t)N_LIT * 64;
    float*     degB = maxB + (size_t)N_LIT * 64;
    _Float16*  w1c  = (_Float16*)(degB + N_LIT);
    _Float16*  w2c  = w1c + 8 * 8  * 32 * 16;   // K=256 -> 8 k-tiles
    _Float16*  w1l  = w2c + 8 * 4  * 32 * 16;   // K=128 -> 4 k-tiles
    _Float16*  w2l  = w1l + 8 * 12 * 32 * 16;   // K=384 -> 12 k-tiles

    // Pack weights into WMMA B-fragment order (once per launch; tiny).
    pack_w<<<(8 * 8  * 512 + 255) / 256, 256, 0, stream>>>(cls_w1, w1c, 256);
    pack_w<<<(8 * 4  * 512 + 255) / 256, 256, 0, stream>>>(cls_w2, w2c, 128);
    pack_w<<<(8 * 12 * 512 + 255) / 256, 256, 0, stream>>>(lit_w1, w1l, 384);
    pack_w<<<(8 * 4  * 512 + 255) / 256, 256, 0, stream>>>(lit_w2, w2l, 128);

    const float NEG_BIG = -3.402823466e38f;

    // ---- Phase A: lit -> cls aggregation ----
    fill_f32x4<<<(N_CLS * 16 + 255) / 256, 256, 0, stream>>>((float4*)sumB, 0.0f, N_CLS * 16);
    fill_f32x4<<<(N_CLS * 16 + 255) / 256, 256, 0, stream>>>((float4*)maxB, NEG_BIG, N_CLS * 16);
    fill_f32x4<<<(N_CLS / 4 + 255) / 256, 256, 0, stream>>>((float4*)degB, 0.0f, N_CLS / 4);
    scatter_agg<<<(unsigned)((long long)NEDGE * 32 / 256), 256, 0, stream>>>(
        h_lit, edge_lit, edge_cls, sumB, maxB, degB);

    // ---- Phase B: clause update (WMMA fused MLP+LN) ----
    mlp_ln<256, true><<<N_CLS / 16, 256, 0, stream>>>(
        h_cls, sumB, maxB, degB, w1c, w2c, cls_b1, cls_b2, cls_gamma, cls_beta, out_cls);

    // ---- Phase C: cls_new -> lit aggregation ----
    fill_f32x4<<<(N_LIT * 16 + 255) / 256, 256, 0, stream>>>((float4*)sumB, 0.0f, N_LIT * 16);
    fill_f32x4<<<(N_LIT * 16 + 255) / 256, 256, 0, stream>>>((float4*)maxB, NEG_BIG, N_LIT * 16);
    fill_f32x4<<<(N_LIT / 4 + 255) / 256, 256, 0, stream>>>((float4*)degB, 0.0f, N_LIT / 4);
    scatter_agg<<<(unsigned)((long long)NEDGE * 32 / 256), 256, 0, stream>>>(
        out_cls, edge_cls, edge_lit, sumB, maxB, degB);

    // ---- Phase D: literal update (WMMA fused MLP+LN) ----
    mlp_ln<384, false><<<N_LIT / 16, 256, 0, stream>>>(
        h_lit, sumB, maxB, degB, w1l, w2l, lit_b1, lit_b2, lit_gamma, lit_beta, out_lit);
}
